// RWKVCell_38843684225192
// MI455X (gfx1250) — matrix operand order repaired and checked by reading, not compiled, not run on hardware
//
#include <hip/hip_runtime.h>

// ---------------------------------------------------------------------------
// RWKV cell forward for MI455X (gfx1250, wave32, WMMA).
// B=4096, E=2048.  GEMMs: bf16 WMMA (v_wmma_f32_16x16x32_bf16), f32 accum,
// double-buffered LDS staged with GLOBAL_LOAD_ASYNC_TO_LDS_B128 (async DMA,
// ASYNCcnt-tracked), branch-free steady-state K loop, compile-time shapes.
// ---------------------------------------------------------------------------

#define E_DIM  2048
#define B_DIM  4096
#define FE_DIM 8192   // 4*E

typedef __attribute__((ext_vector_type(16))) __bf16         v16bf;
typedef __attribute__((ext_vector_type(8)))  float          v8f;
typedef __attribute__((ext_vector_type(4)))  unsigned int   u32x4;
typedef __attribute__((ext_vector_type(4)))  unsigned short u16x4;
typedef int v4i_gcc __attribute__((vector_size(16)));       // matches builtin

union ABFrag { u32x4 u4[2]; v16bf v; };

__device__ __forceinline__ unsigned short f2bf(float f) {
  unsigned int u = __float_as_uint(f);
  u += 0x7FFFu + ((u >> 16) & 1u);   // round-to-nearest-even
  return (unsigned short)(u >> 16);
}
__device__ __forceinline__ float clip50(float v) {
  return fminf(fmaxf(v, -50.f), 50.f);
}

// ---- async global->LDS staging (CDNA5) ------------------------------------
// builtin signature (probe-verified): (v4i addrspace(1)*, v4i addrspace(3)*,
//                                      imm int offset, imm int cpol)
#if defined(__gfx1250__) && __has_builtin(__builtin_amdgcn_global_load_async_to_lds_b128)
#define HAVE_ASYNC_LDS 1
#define COPY16(g, l)                                                          \
  __builtin_amdgcn_global_load_async_to_lds_b128(                             \
      (__attribute__((address_space(1))) v4i_gcc*)(unsigned long long)(const void*)(g), \
      (__attribute__((address_space(3))) v4i_gcc*)(unsigned int)(unsigned long long)(const void*)(l), \
      0, 0)
#define WAIT_ASYNC_STEADY() asm volatile("s_wait_asynccnt 6" ::: "memory")
#define WAIT_ASYNC_ALL()    asm volatile("s_wait_asynccnt 0" ::: "memory")
#else
#define HAVE_ASYNC_LDS 0
#define COPY16(g, l) (*(u32x4*)(l) = *(const u32x4*)(g))
#define WAIT_ASYNC_STEADY()
#define WAIT_ASYNC_ALL()
#endif

// ---------------------------------------------------------------------------
// fp32 -> bf16 conversion (weights / activations), 4-wide
// ---------------------------------------------------------------------------
__global__ __launch_bounds__(256)
void cvt_f32_to_bf16(const float* __restrict__ src,
                     unsigned short* __restrict__ dst, int n4) {
  int i = blockIdx.x * blockDim.x + threadIdx.x;
  if (i >= n4) return;
  const float4 f = ((const float4*)src)[i];
  u16x4 o;
  o.x = f2bf(f.x); o.y = f2bf(f.y); o.z = f2bf(f.z); o.w = f2bf(f.w);
  ((u16x4*)dst)[i] = o;
}

// ---------------------------------------------------------------------------
// bf16 NT-GEMM:  out[m,n] = epilogue( sum_k A[m,k] * W[n,k] )
// A: [B_DIM,K] bf16 row-major, W: [N,K] bf16 row-major (i.e. act @ W.T).
// Block tile 256(M) x 128(N), 256 threads = 8 wave32 in a 4x2 grid,
// each wave computes 64x64 (4x4 WMMA tiles, 16 wmma / K-step).
// K staged in BK=32 slices, double-buffered LDS, async copies,
// branch-free steady-state loop (last slice peeled).
// ---------------------------------------------------------------------------
#define BLK_M 256
#define BLK_N 128
#define BK    32
#define LDT   40   // padded LDS leading dim in halfwords (80B, 16B multiple)

enum { EPI_F32 = 0, EPI_SIGMOID = 1, EPI_KMIN = 2,
       EPI_RELU2_BF16 = 3, EPI_RESID = 4, EPI_RESID_RMUL = 5 };

template<int EPI, int NDIM, int KDIM>
__global__ __launch_bounds__(256)
void gemm_bf16_nt(const unsigned short* __restrict__ Amat,
                  const unsigned short* __restrict__ Wmat,
                  void* __restrict__ outp,
                  const float* __restrict__ aux0,
                  const float* __restrict__ aux1)
{
  __shared__ __align__(16) unsigned short SA[2][BLK_M * LDT];
  __shared__ __align__(16) unsigned short SB[2][BLK_N * LDT];

  const int tid  = threadIdx.x;
  const int lane = tid & 31;
  const int wave = tid >> 5;
  const int wm   = wave >> 1;        // 0..3  (M direction, 64 rows each)
  const int wn   = wave & 1;         // 0..1  (N direction, 64 cols each)
  const int lm   = lane & 15;
  const int lhi  = lane >> 4;
  const int mBase = blockIdx.y * BLK_M;
  const int nBase = blockIdx.x * BLK_N;

  v8f acc[4][4] = {};

  // stage one BK-slice of A (256x32) and W (128x32) into LDS buffer `b`
  auto stage = [&](int b, int k0) {
#pragma unroll
    for (int i = 0; i < 4; ++i) {            // A: 1024 16B-chunks / 256 thr
      int c = tid + (i << 8);
      int row = c >> 2, col = (c & 3) << 3;
      COPY16(&Amat[(size_t)(mBase + row) * KDIM + k0 + col],
             &SA[b][row * LDT + col]);
    }
#pragma unroll
    for (int i = 0; i < 2; ++i) {            // W: 512 16B-chunks / 256 thr
      int c = tid + (i << 8);
      int row = c >> 2, col = (c & 3) << 3;
      COPY16(&Wmat[(size_t)(nBase + row) * KDIM + k0 + col],
             &SB[b][row * LDT + col]);
    }
  };

  // 16 WMMAs on the slice in LDS buffer `b`
  auto compute = [&](int b) {
    // A fragment (16x32): lanes 0-15 M, K={0..7,16..23}; lanes 16-31
    // K={8..15,24..31} -> 16B chunks at halfword cols lhi*8 and lhi*8+16.
    ABFrag a[4];
#pragma unroll
    for (int tm = 0; tm < 4; ++tm) {
      const unsigned short* p = &SA[b][(wm * 64 + tm * 16 + lm) * LDT + lhi * 8];
      a[tm].u4[0] = *(const u32x4*)p;
      a[tm].u4[1] = *(const u32x4*)(p + 16);
    }
    // B fragment (32x16): lane = N, lanes 0-15 K={0..7,16..23}; lanes 16-31
    // K={8..15,24..31} -> chunks at cols lhi*8 and lhi*8+16.
    ABFrag bfr[4];
#pragma unroll
    for (int tn = 0; tn < 4; ++tn) {
      const unsigned short* p = &SB[b][(wn * 64 + tn * 16 + lm) * LDT + lhi * 8];
      bfr[tn].u4[0] = *(const u32x4*)p;
      bfr[tn].u4[1] = *(const u32x4*)(p + 16);
    }
#pragma unroll
    for (int tm = 0; tm < 4; ++tm)
#pragma unroll
      for (int tn = 0; tn < 4; ++tn)
        acc[tm][tn] = __builtin_amdgcn_wmma_f32_16x16x32_bf16(
            false, a[tm].v, false, bfr[tn].v, (short)0, acc[tm][tn],
            false, false);
  };

  stage(0, 0);
  int buf = 0;
#pragma unroll 1
  for (int k0 = 0; k0 < KDIM - BK; k0 += BK) {   // steady state: branch-free
    stage(buf ^ 1, k0 + BK);                     // prefetch next slice (async)
    WAIT_ASYNC_STEADY();                         // retire current slice's 6
    __syncthreads();
    compute(buf);
    __syncthreads();                             // all done reading `buf`
    buf ^= 1;
  }
  WAIT_ASYNC_ALL();                              // peeled last slice
  __syncthreads();
  compute(buf);

  // epilogue: C/D layout -> vgpr r: M = lhi*8 + r, N = lm
  float*          outf = (float*)outp;
  unsigned short* outh = (unsigned short*)outp;
#pragma unroll
  for (int tm = 0; tm < 4; ++tm)
#pragma unroll
    for (int tn = 0; tn < 4; ++tn)
#pragma unroll
      for (int r = 0; r < 8; ++r) {
        int gm = mBase + wm * 64 + tm * 16 + lhi * 8 + r;
        int gn = nBase + wn * 64 + tn * 16 + lm;
        size_t idx = (size_t)gm * NDIM + gn;
        float v = acc[tm][tn][r];
        if (EPI == EPI_SIGMOID) {
          outf[idx] = 1.f / (1.f + __expf(-v));
        } else if (EPI == EPI_KMIN) {
          outf[idx] = fminf(v, 60.f);
        } else if (EPI == EPI_RELU2_BF16) {
          float t = fmaxf(v, 0.f);
          outh[idx] = f2bf(t * t);
        } else if (EPI == EPI_RESID) {
          outf[idx] = aux0[idx] + v;
        } else if (EPI == EPI_RESID_RMUL) {
          outf[idx] = aux0[idx] + aux1[idx] * v;
        } else {
          outf[idx] = v;
        }
      }
}

// ---------------------------------------------------------------------------
// LayerNorm(x) + time-mix inputs (bf16) + state ch0.  One block per row.
// ---------------------------------------------------------------------------
__global__ __launch_bounds__(256)
void ln_mix1(const float* __restrict__ x, const float* __restrict__ state,
             const float* __restrict__ g, const float* __restrict__ be,
             const float* __restrict__ tmk, const float* __restrict__ tmv,
             const float* __restrict__ tmr,
             unsigned short* __restrict__ xk, unsigned short* __restrict__ xv,
             unsigned short* __restrict__ xr, float* __restrict__ st_out)
{
  const int brow = blockIdx.x;
  const int tid  = threadIdx.x;
  const float* xrow = x + (size_t)brow * E_DIM;
  __shared__ float rs[256], rq[256];
  float s = 0.f, q = 0.f;
  for (int e = tid; e < E_DIM; e += 256) { float t = xrow[e]; s += t; q += t * t; }
  rs[tid] = s; rq[tid] = q; __syncthreads();
  for (int o = 128; o > 0; o >>= 1) {
    if (tid < o) { rs[tid] += rs[tid + o]; rq[tid] += rq[tid + o]; }
    __syncthreads();
  }
  const float mu   = rs[0] * (1.f / E_DIM);
  const float var  = rq[0] * (1.f / E_DIM) - mu * mu;
  const float rstd = rsqrtf(var + 1e-5f);
  for (int e = tid; e < E_DIM; e += 256) {
    float xn = (xrow[e] - mu) * rstd * g[e] + be[e];
    size_t si = ((size_t)brow * E_DIM + e) * 5;
    size_t oi = (size_t)brow * E_DIM + e;
    float sx = state[si + 0];
    float tk = tmk[e], tv = tmv[e], tr = tmr[e];
    xk[oi] = f2bf(xn * tk + sx * (1.f - tk));
    xv[oi] = f2bf(xn * tv + sx * (1.f - tv));
    xr[oi] = f2bf(xn * tr + sx * (1.f - tr));
    st_out[si + 0] = clip50(xn);
  }
}

// ---------------------------------------------------------------------------
// WKV exponential recurrence (single step), states ch1..3, rwkv = r*wkv (bf16)
// ---------------------------------------------------------------------------
__global__ __launch_bounds__(256)
void wkv_step(const float* __restrict__ r, const float* __restrict__ k,
              const float* __restrict__ v, const float* __restrict__ state,
              const float* __restrict__ tdec, const float* __restrict__ tfirst,
              unsigned short* __restrict__ rwkv, float* __restrict__ st_out)
{
  size_t i = (size_t)blockIdx.x * blockDim.x + threadIdx.x;
  if (i >= (size_t)B_DIM * E_DIM) return;
  const int e = (int)(i & (E_DIM - 1));
  const size_t si = i * 5;
  float aa = state[si + 1], bb = state[si + 2], pp = state[si + 3];
  float kk = k[i], vv = v[i];
  float ww  = fminf(tfirst[e] + kk, 30.f);
  float ppc = fminf(pp, 30.f);
  float p   = fmaxf(ppc, ww);
  float e1  = __expf(ppc - p), e2 = __expf(ww - p);
  float wkv = (e1 * aa + e2 * vv) / (e1 * bb + e2 + 1e-8f);
  float ww2 = fminf(ppc + tdec[e], 30.f);
  float p2  = fmaxf(ww2, kk);
  float e1b = __expf(ww2 - p2), e2b = __expf(kk - p2);
  st_out[si + 1] = clip50(e1b * aa + e2b * vv);
  st_out[si + 2] = clip50(e1b * bb + e2b);
  st_out[si + 3] = clip50(p2);
  rwkv[i] = f2bf(r[i] * wkv);
}

// ---------------------------------------------------------------------------
// LayerNorm2(x1) + channel-mix inputs (bf16) + state ch4
// ---------------------------------------------------------------------------
__global__ __launch_bounds__(256)
void ln_mix2(const float* __restrict__ x1, const float* __restrict__ state,
             const float* __restrict__ g, const float* __restrict__ be,
             const float* __restrict__ tkcm, const float* __restrict__ trcm,
             unsigned short* __restrict__ xk2, unsigned short* __restrict__ xr2,
             float* __restrict__ st_out)
{
  const int brow = blockIdx.x;
  const int tid  = threadIdx.x;
  const float* xrow = x1 + (size_t)brow * E_DIM;
  __shared__ float rs[256], rq[256];
  float s = 0.f, q = 0.f;
  for (int e = tid; e < E_DIM; e += 256) { float t = xrow[e]; s += t; q += t * t; }
  rs[tid] = s; rq[tid] = q; __syncthreads();
  for (int o = 128; o > 0; o >>= 1) {
    if (tid < o) { rs[tid] += rs[tid + o]; rq[tid] += rq[tid + o]; }
    __syncthreads();
  }
  const float mu   = rs[0] * (1.f / E_DIM);
  const float var  = rq[0] * (1.f / E_DIM) - mu * mu;
  const float rstd = rsqrtf(var + 1e-5f);
  for (int e = tid; e < E_DIM; e += 256) {
    float xn = (xrow[e] - mu) * rstd * g[e] + be[e];
    size_t si = ((size_t)brow * E_DIM + e) * 5;
    size_t oi = (size_t)brow * E_DIM + e;
    float sx = state[si + 4];
    float tk = tkcm[e], tr = trcm[e];
    xk2[oi] = f2bf(xn * tk + sx * (1.f - tk));
    xr2[oi] = f2bf(xn * tr + sx * (1.f - tr));
    st_out[si + 4] = clip50(xn);
  }
}

// ---------------------------------------------------------------------------
// host side
// ---------------------------------------------------------------------------
template<int EPI, int NDIM, int KDIM>
static void launch_gemm(const unsigned short* A, const unsigned short* W,
                        void* out, const float* aux0, const float* aux1,
                        hipStream_t s) {
  dim3 grid(NDIM / BLK_N, B_DIM / BLK_M);
  gemm_bf16_nt<EPI, NDIM, KDIM><<<grid, 256, 0, s>>>(A, W, out, aux0, aux1);
}

static void launch_cvt(const float* src, unsigned short* dst, size_t n,
                       hipStream_t s) {
  int n4 = (int)(n / 4);
  cvt_f32_to_bf16<<<(n4 + 255) / 256, 256, 0, s>>>(src, dst, n4);
}

extern "C" void kernel_launch(void* const* d_in, const int* in_sizes, int n_in,
                              void* d_out, int out_size, void* d_ws, size_t ws_size,
                              hipStream_t stream) {
  (void)in_sizes; (void)n_in; (void)out_size; (void)ws_size;

  const float* x          = (const float*)d_in[0];
  const float* state      = (const float*)d_in[1];
  const float* ln1_g      = (const float*)d_in[2];
  const float* ln1_b      = (const float*)d_in[3];
  const float* ln2_g      = (const float*)d_in[4];
  const float* ln2_b      = (const float*)d_in[5];
  const float* time_decay = (const float*)d_in[6];
  const float* time_first = (const float*)d_in[7];
  const float* tm_k       = (const float*)d_in[8];
  const float* tm_v       = (const float*)d_in[9];
  const float* tm_r       = (const float*)d_in[10];
  const float* tm_k_cm    = (const float*)d_in[11];
  const float* tm_r_cm    = (const float*)d_in[12];
  const float* Wk         = (const float*)d_in[13];
  const float* Wv         = (const float*)d_in[14];
  const float* Wr         = (const float*)d_in[15];
  const float* Wo         = (const float*)d_in[16];
  const float* Wk_cm      = (const float*)d_in[17];
  const float* Wr_cm      = (const float*)d_in[18];
  const float* Wv_cm      = (const float*)d_in[19];

  float* out      = (float*)d_out;                 // x_out [B,E]
  float* st_out   = out + (size_t)B_DIM * E_DIM;   // new_state [B,E,5]

  unsigned char* cur = (unsigned char*)d_ws;
  auto take = [&](size_t bytes) -> unsigned char* {
    unsigned char* p = cur; cur += bytes; return p;
  };
  const size_t EE  = (size_t)E_DIM * E_DIM;
  const size_t BE  = (size_t)B_DIM * E_DIM;
  const size_t BFE = (size_t)B_DIM * FE_DIM;

  unsigned short* Wk_bf   = (unsigned short*)take(EE * 2);
  unsigned short* Wv_bf   = (unsigned short*)take(EE * 2);
  unsigned short* Wr_bf   = (unsigned short*)take(EE * 2);
  unsigned short* Wo_bf   = (unsigned short*)take(EE * 2);
  unsigned short* Wrcm_bf = (unsigned short*)take(EE * 2);
  unsigned short* Wkcm_bf = (unsigned short*)take(EE * 4 * 2);
  unsigned short* Wvcm_bf = (unsigned short*)take(EE * 4 * 2);
  unsigned short* xk_bf   = (unsigned short*)take(BE * 2);
  unsigned short* xv_bf   = (unsigned short*)take(BE * 2);
  unsigned short* xr_bf   = (unsigned short*)take(BE * 2);
  float*          r_f     = (float*)take(BE * 4);
  float*          k_f     = (float*)take(BE * 4);
  float*          v_f     = (float*)take(BE * 4);
  unsigned short* rwkv_bf = (unsigned short*)take(BE * 2);
  float*          x1_f    = (float*)take(BE * 4);
  unsigned short* xk2_bf  = (unsigned short*)take(BE * 2);
  unsigned short* xr2_bf  = (unsigned short*)take(BE * 2);
  float*          r2_f    = (float*)take(BE * 4);
  unsigned short* k2_bf   = (unsigned short*)take(BFE * 2);

  // 1) weights fp32 -> bf16
  launch_cvt(Wk,    Wk_bf,   EE,     stream);
  launch_cvt(Wv,    Wv_bf,   EE,     stream);
  launch_cvt(Wr,    Wr_bf,   EE,     stream);
  launch_cvt(Wo,    Wo_bf,   EE,     stream);
  launch_cvt(Wr_cm, Wrcm_bf, EE,     stream);
  launch_cvt(Wk_cm, Wkcm_bf, EE * 4, stream);
  launch_cvt(Wv_cm, Wvcm_bf, EE * 4, stream);

  // 2) LN1 + time-mix
  ln_mix1<<<B_DIM, 256, 0, stream>>>(x, state, ln1_g, ln1_b, tm_k, tm_v, tm_r,
                                     xk_bf, xv_bf, xr_bf, st_out);

  // 3) r / k / v projections
  launch_gemm<EPI_SIGMOID, E_DIM, E_DIM>(xr_bf, Wr_bf, r_f, nullptr, nullptr, stream);
  launch_gemm<EPI_KMIN,    E_DIM, E_DIM>(xk_bf, Wk_bf, k_f, nullptr, nullptr, stream);
  launch_gemm<EPI_F32,     E_DIM, E_DIM>(xv_bf, Wv_bf, v_f, nullptr, nullptr, stream);

  // 4) WKV recurrence
  wkv_step<<<(int)(BE / 256), 256, 0, stream>>>(r_f, k_f, v_f, state,
                                                time_decay, time_first,
                                                rwkv_bf, st_out);

  // 5) x1 = x + (r*wkv) @ Wo.T
  launch_gemm<EPI_RESID, E_DIM, E_DIM>(rwkv_bf, Wo_bf, x1_f, x, nullptr, stream);

  // 6) LN2 + channel-mix
  ln_mix2<<<B_DIM, 256, 0, stream>>>(x1_f, state, ln2_g, ln2_b,
                                     tm_k_cm, tm_r_cm, xk2_bf, xr2_bf, st_out);

  // 7) r2 and k2
  launch_gemm<EPI_SIGMOID,    E_DIM,  E_DIM>(xr2_bf, Wrcm_bf, r2_f, nullptr, nullptr, stream);
  launch_gemm<EPI_RELU2_BF16, FE_DIM, E_DIM>(xk2_bf, Wkcm_bf, k2_bf, nullptr, nullptr, stream);

  // 8) x_out = x1 + r2 * (k2 @ Wv_cm.T)
  launch_gemm<EPI_RESID_RMUL, E_DIM, FE_DIM>(k2_bf, Wvcm_bf, out, x1_f, r2_f, stream);
}
